// TransformerBlock_42640435315372
// MI455X (gfx1250) — matrix operand order repaired and checked by reading, not compiled, not run on hardware
//
#include <hip/hip_runtime.h>
#include <hip/hip_bf16.h>

// ---------------------------------------------------------------------------
// Transformer block for MI455X (gfx1250, wave32).
// - 6 GEMMs on v_wmma_f32_16x16x32_bf16 (fp32 accumulate)
// - Operands pre-packed to bf16 (weights pre-transposed to [N][K])
// - Tiles staged via Tensor Data Mover (tensor_load_to_lds + s_wait_tensorcnt)
//   into double-buffered LDS with hardware padding (pitch 40 halves)
// - Softmax / LayerNorm / residuals in fp32
// ---------------------------------------------------------------------------

typedef __bf16 bf16_t;
typedef __bf16 v16bf __attribute__((ext_vector_type(16)));
typedef __bf16 v8bf  __attribute__((ext_vector_type(8)));
typedef float  v8f   __attribute__((ext_vector_type(8)));
typedef unsigned int u32x4 __attribute__((ext_vector_type(4)));
typedef int          i32x8 __attribute__((ext_vector_type(8)));
typedef int          i32x4 __attribute__((ext_vector_type(4)));

#define DMODEL 1024
#define DFF    4096

// GEMM tiling
#define BM  128
#define BN  128
#define BK  32
#define LDP 40   // LDS pitch in bf16 elems: 32 data + 8 pad (= TDM pad 16 DW + 4 DW)

enum { EPI_BF16 = 0, EPI_RES_F32 = 1, EPI_BIAS_RELU_BF16 = 2, EPI_BIAS_RES_F32 = 3 };

union V16U { v16bf v; v8bf h[2]; };

#if defined(__has_builtin)
#if __has_builtin(__builtin_amdgcn_tensor_load_to_lds)
#define HAS_TDM 1
#endif
#endif
#ifndef HAS_TDM
#define HAS_TDM 0
#endif

#if HAS_TDM
// ---- Tensor DMA descriptor (CDNA5 ISA ch.8) for a 2D bf16 tile: ------------
//   tile_dim0 = 32 elems (64B row), tile_dim1 = 128 rows, row stride = rowLen,
//   LDS padding: +4 DWORDs after every 16 DWORDs  ->  pitch 40 bf16 elems.
__device__ __forceinline__ i32x8 tdm_group1(int rowLen, int rowsTotal)
{
    i32x8 g1;
    g1[0] = (1 << 16)            // data_size = 1 (2 bytes)
          | (1 << 20)            // pad_enable
          | (3 << 22)            // pad_interval: code 3 = 16 DWORDs
          | (3 << 25);           // pad_amount:   code 3 = 4 DWORDs
    g1[1] = (rowLen & 0xFFFF) << 16;                                  // tensor_dim0 lo
    g1[2] = ((rowLen >> 16) & 0xFFFF) | ((rowsTotal & 0xFFFF) << 16); // dim0 hi | dim1 lo
    g1[3] = ((rowsTotal >> 16) & 0xFFFF) | (BK << 16);                // dim1 hi | tile_dim0
    g1[4] = BM;                                                       // tile_dim1 | tile_dim2=0
    g1[5] = rowLen;                                                   // tensor_dim0_stride lo
    g1[6] = 0;                                                        // stride hi | dim1_stride lo
    g1[7] = 0;
    return g1;
}

__device__ __forceinline__ void tdm_issue(unsigned ldsOff, const bf16_t* gp, i32x8 g1)
{
    unsigned long long ga = (unsigned long long)(uintptr_t)gp;
    u32x4 g0;
    g0[0] = 1u;                                          // count=1, user mode
    g0[1] = ldsOff;                                      // lds_addr
    g0[2] = (unsigned)ga;                                // global_addr lo
    g0[3] = (unsigned)((ga >> 32) & 0x01FFFFFFull) | 0x80000000u; // addr hi | type=2
    const i32x4 z4 = {0, 0, 0, 0};
    const i32x8 z8 = {0, 0, 0, 0, 0, 0, 0, 0};
    // clang-23 / therock-10.0 6-arg form
    __builtin_amdgcn_tensor_load_to_lds(g0, g1, z4, z4, z8, 0);
}
#endif

// ---- fragment loads + 8 WMMAs on one staged tile pair ----------------------
__device__ __forceinline__ void mma_tile(const bf16_t* As, const bf16_t* Bs,
                                         int wr, int wc, int half, int l16,
                                         v8f (&acc)[4][2])
{
    // A 16x32 bf16 layout: lane<16 -> K 0..7,16..23 of row l16; lane>=16 -> K 8..15,24..31
    // B 32x16 bf16 layout: lane<16 -> K 0..7,16..23 of col l16; lane>=16 -> K 8..15,24..31
    v16bf afr[4], bfr[2];
#pragma unroll
    for (int mt = 0; mt < 4; ++mt) {
        const int r = wr + mt * 16 + l16;
        V16U u;
        u.h[0] = *(const v8bf*)&As[r * LDP + half * 8];
        u.h[1] = *(const v8bf*)&As[r * LDP + 16 + half * 8];
        afr[mt] = u.v;
    }
#pragma unroll
    for (int nt = 0; nt < 2; ++nt) {
        const int c = wc + nt * 16 + l16;
        V16U u;
        u.h[0] = *(const v8bf*)&Bs[c * LDP + half * 8];
        u.h[1] = *(const v8bf*)&Bs[c * LDP + 16 + half * 8];
        bfr[nt] = u.v;
    }
#pragma unroll
    for (int mt = 0; mt < 4; ++mt)
#pragma unroll
        for (int nt = 0; nt < 2; ++nt)
            acc[mt][nt] = __builtin_amdgcn_wmma_f32_16x16x32_bf16(
                false, afr[mt], false, bfr[nt], (short)0, acc[mt][nt], false, false);
}

#if !HAS_TDM
// fallback staging: pure b128 copies (operands already bf16, already transposed)
__device__ __forceinline__ void stage_copy(bf16_t* dst, const bf16_t* src, int ld, int tid)
{
#pragma unroll
    for (int c = 0; c < 2; ++c) {
        const int idx = c * 256 + tid;          // 512 x 16B chunks
        const int m  = idx >> 2;
        const int k8 = (idx & 3) * 8;
        const uint4 v = *(const uint4*)(src + (size_t)m * ld + k8);
        *(uint4*)(dst + m * LDP + k8) = v;
    }
}
#endif

// ---------------------------------------------------------------------------
// C[M,NC] = A[M,K] @ Bt[NC,K]^T   (A,Bt bf16; fp32 acc; fused epilogues)
// 256 threads = 8 waves (2x4), wave tile 64x32, block tile 128x128, BK=32.
// ---------------------------------------------------------------------------
template<int EPI>
__global__ __launch_bounds__(256)
void gemm_tdm_kernel(const bf16_t* __restrict__ A, const bf16_t* __restrict__ Bt,
                     const float* __restrict__ bias, const float* __restrict__ resid,
                     void* __restrict__ outp, int M, int NC, int K)
{
    __shared__ __align__(16) bf16_t lds[2 * (BM + BN) * LDP];

    const int tid  = threadIdx.x;
    const int l    = tid & 31;
    const int half = l >> 4;
    const int l16  = l & 15;
    const int w    = tid >> 5;
    const int wr   = (w >> 2) * 64;
    const int wc   = (w & 3) * 32;

    const int tileM = blockIdx.y * BM;
    const int tileN = blockIdx.x * BN;
    const bf16_t* gA = A  + (size_t)tileM * K;
    const bf16_t* gB = Bt + (size_t)tileN * K;

    v8f acc[4][2];
    const v8f vzero = {0.f, 0.f, 0.f, 0.f, 0.f, 0.f, 0.f, 0.f};
#pragma unroll
    for (int i = 0; i < 4; ++i)
#pragma unroll
        for (int j = 0; j < 2; ++j) acc[i][j] = vzero;

    const int kTiles = K / BK;

#if HAS_TDM
    const int wu = __builtin_amdgcn_readfirstlane(w);   // scalar branch -> one issue/wave
    const unsigned ldsBase   = (unsigned)(uintptr_t)&lds[0];
    const unsigned bufStride = (unsigned)((BM + BN) * LDP * sizeof(bf16_t));
    const unsigned bOff      = (unsigned)(BM * LDP * sizeof(bf16_t));
    const i32x8 g1A = tdm_group1(K, M);
    const i32x8 g1B = tdm_group1(K, NC);

    if (wu == 0)      tdm_issue(ldsBase,        gA, g1A);
    else if (wu == 1) tdm_issue(ldsBase + bOff, gB, g1B);
    if (wu < 2) __builtin_amdgcn_s_wait_tensorcnt(0);
    __syncthreads();

    for (int kt = 0; kt < kTiles; ++kt) {
        const int cur = kt & 1;
        const bf16_t* As = lds + cur * (BM + BN) * LDP;
        const bf16_t* Bs = As + BM * LDP;

        if (kt + 1 < kTiles) {      // DMA next tile while computing this one
            const unsigned nb = ldsBase + (unsigned)(cur ^ 1) * bufStride;
            if (wu == 0)      tdm_issue(nb,        gA + (size_t)(kt + 1) * BK, g1A);
            else if (wu == 1) tdm_issue(nb + bOff, gB + (size_t)(kt + 1) * BK, g1B);
        }

        mma_tile(As, Bs, wr, wc, half, l16, acc);

        if (wu < 2 && kt + 1 < kTiles) __builtin_amdgcn_s_wait_tensorcnt(0);
        __syncthreads();
    }
#else
    for (int kt = 0; kt < kTiles; ++kt) {
        bf16_t* As = lds;
        bf16_t* Bs = lds + BM * LDP;
        stage_copy(As, gA + (size_t)kt * BK, K, tid);
        stage_copy(Bs, gB + (size_t)kt * BK, K, tid);
        __syncthreads();
        mma_tile(As, Bs, wr, wc, half, l16, acc);
        __syncthreads();
    }
#endif

    // ---- epilogue: D layout -> VGPR g holds M = g + 8*(lane>=16), N = lane&15
#pragma unroll
    for (int mt = 0; mt < 4; ++mt) {
#pragma unroll
        for (int nt = 0; nt < 2; ++nt) {
            const int col  = tileN + wc + nt * 16 + l16;
            const int row0 = tileM + wr + mt * 16 + half * 8;
#pragma unroll
            for (int g = 0; g < 8; ++g) {
                const int row = row0 + g;
                const float vacc = acc[mt][nt][g];
                const size_t o = (size_t)row * NC + col;
                if constexpr (EPI == EPI_BF16) {
                    ((bf16_t*)outp)[o] = (bf16_t)vacc;
                } else if constexpr (EPI == EPI_RES_F32) {
                    ((float*)outp)[o] = vacc + resid[o];
                } else if constexpr (EPI == EPI_BIAS_RELU_BF16) {
                    const float y = vacc + bias[col];
                    ((bf16_t*)outp)[o] = (bf16_t)fmaxf(y, 0.f);
                } else { // EPI_BIAS_RES_F32
                    ((float*)outp)[o] = vacc + bias[col] + resid[o];
                }
            }
        }
    }
}

// ---------------------------------------------------------------------------
// fp32 -> bf16 pack (activations). 4 elems / thread.
// ---------------------------------------------------------------------------
__global__ __launch_bounds__(256)
void pack_bf16_kernel(const float* __restrict__ in, bf16_t* __restrict__ out)
{
    const size_t i = ((size_t)blockIdx.x * 256 + threadIdx.x) * 4;
    const float4 f = *(const float4*)(in + i);
    out[i + 0] = (bf16_t)f.x;
    out[i + 1] = (bf16_t)f.y;
    out[i + 2] = (bf16_t)f.z;
    out[i + 3] = (bf16_t)f.w;
}

// ---------------------------------------------------------------------------
// Weight pack: W[K][N] fp32 -> Wt[N][K] bf16 (64x64 LDS tile transpose).
// grid = (N/64, K/64)
// ---------------------------------------------------------------------------
__global__ __launch_bounds__(256)
void packT_bf16_kernel(const float* __restrict__ W, bf16_t* __restrict__ Wt,
                       int K, int N)
{
    __shared__ float t[64][65];
    const int tid = threadIdx.x;
    const int n0 = blockIdx.x * 64;
    const int k0 = blockIdx.y * 64;
#pragma unroll
    for (int i = 0; i < 16; ++i) {
        const int idx = i * 256 + tid;
        const int r = idx >> 6, c = idx & 63;
        t[r][c] = W[(size_t)(k0 + r) * N + n0 + c];
    }
    __syncthreads();
#pragma unroll
    for (int i = 0; i < 16; ++i) {
        const int idx = i * 256 + tid;
        const int r = idx >> 6, c = idx & 63;          // r: out row (n), c: out col (k)
        Wt[(size_t)(n0 + r) * K + k0 + c] = (bf16_t)t[c][r];
    }
}

// ---------------------------------------------------------------------------
// Per-token "attention": energy = Qh(64x16)@Kh(16x64)/32, softmax over cols,
// concat = attn(64x64)@Vh(64x16). One 256-thread block per token.
// ---------------------------------------------------------------------------
__global__ __launch_bounds__(256)
void attention_kernel(const bf16_t* __restrict__ Q, const bf16_t* __restrict__ Kk,
                      const bf16_t* __restrict__ V, const int* __restrict__ mask,
                      bf16_t* __restrict__ out)
{
    __shared__ float qs[1024];
    __shared__ float ks[1024];
    __shared__ float vsp[16 * 68];
    __shared__ float enp[64 * 65];

    const int tok = blockIdx.x;
    const int tid = threadIdx.x;
    const size_t base = (size_t)tok * DMODEL;

    {
        const int idx = tid * 4;
        const int h = idx >> 6, a = idx & 63;
#pragma unroll
        for (int j = 0; j < 4; ++j) qs[idx + j] = (float)Q[base + idx + j];
#pragma unroll
        for (int j = 0; j < 4; ++j) ks[idx + j] = (float)Kk[base + idx + j];
#pragma unroll
        for (int j = 0; j < 4; ++j) vsp[h * 68 + a + j] = (float)V[base + idx + j];
    }
    __syncthreads();

    const int mk = mask[tok];
#pragma unroll 4
    for (int i = 0; i < 16; ++i) {
        const int e = i * 256 + tid;
        const int a = e >> 6, b = e & 63;
        float s = 0.f;
#pragma unroll
        for (int h = 0; h < 16; ++h) s += qs[h * 64 + a] * ks[h * 64 + b];
        enp[a * 65 + b] = mk ? s * 0.03125f : 0.f;   // 1/(D_HEAD/2) = 1/32
    }
    __syncthreads();

    if (tid < 64) {
        float mx = -1e30f;
        for (int j = 0; j < 64; ++j) mx = fmaxf(mx, enp[tid * 65 + j]);
        float sum = 0.f;
        for (int j = 0; j < 64; ++j) {
            const float ex = __expf(enp[tid * 65 + j] - mx);
            enp[tid * 65 + j] = ex;
            sum += ex;
        }
        const float inv = 1.f / sum;
        for (int j = 0; j < 64; ++j) enp[tid * 65 + j] *= inv;
    }
    __syncthreads();

#pragma unroll
    for (int i = 0; i < 4; ++i) {
        const int e = i * 256 + tid;
        const int a = e >> 4, h = e & 15;
        float s = 0.f;
#pragma unroll 16
        for (int b = 0; b < 64; ++b) s += enp[a * 65 + b] * vsp[h * 68 + b];
        out[base + e] = (bf16_t)s;   // concat[a*16+h]
    }
}

// ---------------------------------------------------------------------------
// LayerNorm over last dim (1024); one block per row; optional bf16 copy out.
// ---------------------------------------------------------------------------
template<bool WB>
__global__ __launch_bounds__(256)
void layernorm_kernel(const float* __restrict__ x, const float* __restrict__ g,
                      const float* __restrict__ b, float* __restrict__ outf,
                      bf16_t* __restrict__ outb)
{
    __shared__ float rs[256];
    __shared__ float rq[256];
    const int tid = threadIdx.x;
    const size_t base = (size_t)blockIdx.x * DMODEL;

    float v[4];
    float s = 0.f, s2 = 0.f;
#pragma unroll
    for (int j = 0; j < 4; ++j) {
        v[j] = x[base + tid + j * 256];
        s  += v[j];
        s2 += v[j] * v[j];
    }
    rs[tid] = s; rq[tid] = s2;
    __syncthreads();
    for (int off = 128; off > 0; off >>= 1) {
        if (tid < off) { rs[tid] += rs[tid + off]; rq[tid] += rq[tid + off]; }
        __syncthreads();
    }
    const float mean = rs[0] * (1.f / DMODEL);
    const float var  = rq[0] * (1.f / DMODEL) - mean * mean;
    const float rstd = rsqrtf(var + 1e-5f);
#pragma unroll
    for (int j = 0; j < 4; ++j) {
        const int idx = tid + j * 256;
        const float y = (v[j] - mean) * rstd * g[idx] + b[idx];
        outf[base + idx] = y;
        if constexpr (WB) outb[base + idx] = (bf16_t)y;
    }
}

// ---------------------------------------------------------------------------
extern "C" void kernel_launch(void* const* d_in, const int* in_sizes, int n_in,
                              void* d_out, int out_size, void* d_ws, size_t ws_size,
                              hipStream_t stream)
{
    const float* queries = (const float*)d_in[0];
    const float* keys    = (const float*)d_in[1];
    const float* values  = (const float*)d_in[2];
    const int*   mask    = (const int*)d_in[3];
    const float* Wq   = (const float*)d_in[4];
    const float* Wk   = (const float*)d_in[5];
    const float* Wv   = (const float*)d_in[6];
    const float* Wo   = (const float*)d_in[7];
    const float* W1   = (const float*)d_in[8];
    const float* b1   = (const float*)d_in[9];
    const float* W2   = (const float*)d_in[10];
    const float* b2   = (const float*)d_in[11];
    const float* ln1g = (const float*)d_in[12];
    const float* ln1b = (const float*)d_in[13];
    const float* ln2g = (const float*)d_in[14];
    const float* ln2b = (const float*)d_in[15];

    const int NT = in_sizes[3];     // n*t tokens (16384)

    char*  ws  = (char*)d_ws;
    size_t off = 0;
    auto carve = [&](size_t bytes) -> void* {
        void* p = ws + off;
        off += (bytes + 255) & ~(size_t)255;
        return p;
    };
    // H1 region aliases the packed QKV inputs (dead after the QKV GEMMs).
    bf16_t* H1   = (bf16_t*)carve((size_t)NT * DFF * 2);     // relu(x@W1+b1)
    bf16_t* Xq   = H1;
    bf16_t* Xk   = Xq + (size_t)NT * DMODEL;
    bf16_t* Xv   = Xk + (size_t)NT * DMODEL;
    bf16_t* Qb   = (bf16_t*)carve((size_t)NT * DMODEL * 2);
    bf16_t* Kb   = (bf16_t*)carve((size_t)NT * DMODEL * 2);
    bf16_t* Vb   = (bf16_t*)carve((size_t)NT * DMODEL * 2);
    bf16_t* Cb   = (bf16_t*)carve((size_t)NT * DMODEL * 2);  // concat, then x_bf16
    float*  F32A = (float*) carve((size_t)NT * DMODEL * 4);  // attn_out, then y2
    float*  F32B = (float*) carve((size_t)NT * DMODEL * 4);  // x (residual)
    bf16_t* WqT  = (bf16_t*)carve((size_t)DMODEL * DMODEL * 2);
    bf16_t* WkT  = (bf16_t*)carve((size_t)DMODEL * DMODEL * 2);
    bf16_t* WvT  = (bf16_t*)carve((size_t)DMODEL * DMODEL * 2);
    bf16_t* WoT  = (bf16_t*)carve((size_t)DMODEL * DMODEL * 2);
    bf16_t* W1T  = (bf16_t*)carve((size_t)DMODEL * DFF * 2); // [DFF][DMODEL]
    bf16_t* W2T  = (bf16_t*)carve((size_t)DFF * DMODEL * 2); // [DMODEL][DFF]
    (void)ws_size; (void)n_in; (void)out_size;

    const dim3 blk(256);
    const dim3 gDM(DMODEL / BN, NT / BM);
    const dim3 gFF(DFF / BN,    NT / BM);
    const int  packBlocks = (NT * DMODEL) / 1024;

    // one-time packs: activations fp32->bf16, weights fp32->bf16 transposed
    pack_bf16_kernel<<<packBlocks, blk, 0, stream>>>(queries, Xq);
    pack_bf16_kernel<<<packBlocks, blk, 0, stream>>>(keys,    Xk);
    pack_bf16_kernel<<<packBlocks, blk, 0, stream>>>(values,  Xv);
    packT_bf16_kernel<<<dim3(DMODEL/64, DMODEL/64), blk, 0, stream>>>(Wq, WqT, DMODEL, DMODEL);
    packT_bf16_kernel<<<dim3(DMODEL/64, DMODEL/64), blk, 0, stream>>>(Wk, WkT, DMODEL, DMODEL);
    packT_bf16_kernel<<<dim3(DMODEL/64, DMODEL/64), blk, 0, stream>>>(Wv, WvT, DMODEL, DMODEL);
    packT_bf16_kernel<<<dim3(DMODEL/64, DMODEL/64), blk, 0, stream>>>(Wo, WoT, DMODEL, DMODEL);
    packT_bf16_kernel<<<dim3(DFF/64,    DMODEL/64), blk, 0, stream>>>(W1, W1T, DMODEL, DFF);
    packT_bf16_kernel<<<dim3(DMODEL/64, DFF/64),    blk, 0, stream>>>(W2, W2T, DFF, DMODEL);

    // QKV projections (bf16 out)
    gemm_tdm_kernel<EPI_BF16><<<gDM, blk, 0, stream>>>(Xq, WqT, nullptr, nullptr, Qb, NT, DMODEL, DMODEL);
    gemm_tdm_kernel<EPI_BF16><<<gDM, blk, 0, stream>>>(Xk, WkT, nullptr, nullptr, Kb, NT, DMODEL, DMODEL);
    gemm_tdm_kernel<EPI_BF16><<<gDM, blk, 0, stream>>>(Xv, WvT, nullptr, nullptr, Vb, NT, DMODEL, DMODEL);

    // per-token attention -> concat (bf16)
    attention_kernel<<<dim3(NT), blk, 0, stream>>>(Qb, Kb, Vb, mask, Cb);

    // attn_out = concat @ Wo + queries  (fp32)
    gemm_tdm_kernel<EPI_RES_F32><<<gDM, blk, 0, stream>>>(Cb, WoT, nullptr, queries, F32A, NT, DMODEL, DMODEL);

    // x = LN1(attn_out): fp32 -> F32B, bf16 -> Cb
    layernorm_kernel<true><<<dim3(NT), blk, 0, stream>>>(F32A, ln1g, ln1b, F32B, Cb);

    // h1 = relu(x @ W1 + b1)  (bf16)   [H1 aliases Xq/Xk/Xv, now dead]
    gemm_tdm_kernel<EPI_BIAS_RELU_BF16><<<gFF, blk, 0, stream>>>(Cb, W1T, b1, nullptr, H1, NT, DFF, DMODEL);

    // y2 = h1 @ W2 + b2 + x   (fp32)
    gemm_tdm_kernel<EPI_BIAS_RES_F32><<<gDM, blk, 0, stream>>>(H1, W2T, b2, F32B, F32A, NT, DMODEL, DFF);

    // out = LN2(y2)  (fp32)
    layernorm_kernel<false><<<dim3(NT), blk, 0, stream>>>(F32A, ln2g, ln2b, (float*)d_out, nullptr);
}
